// ConvKNRM_49838800503290
// MI455X (gfx1250) — matrix operand order repaired and checked by reading, not compiled, not run on hardware
//
#include <hip/hip_runtime.h>
#include <hip/hip_bf16.h>

// Conv-KNRM for MI455X (gfx1250, wave32).
// - 9 matching GEMMs (512x128)x(128x128) per batch of 128 (~19 GFLOP):
//   v_wmma_f32_16x16x32_bf16 fused with Gaussian soft-histogram pooling
//   (exp2) on the accumulator tiles; doc pane staged via the Tensor Data
//   Mover (tensor_load_to_lds, double-buffered, s_wait_tensorcnt) shared by
//   all 8 waves of a block.
// - Convs (~57 GFLOP) are ALSO WMMA: im2col A-fragments built in LDS in the
//   contiguous-k A layout (k = 16*(lane>=16) + slot), weights pre-repacked as bf16 B-fragment
//   tiles; epilogue does bias+tanh+cross-wave L2-norm and stores outputs
//   pre-swizzled into WMMA A/B fragment layouts for the match kernel.

typedef __attribute__((ext_vector_type(16))) __bf16 v16bf;
typedef __attribute__((ext_vector_type(8)))  float  v8f;
typedef __attribute__((ext_vector_type(4)))  unsigned int v4u;
typedef __attribute__((ext_vector_type(4)))  int v4i;
typedef __attribute__((ext_vector_type(8)))  int v8i;

#define H_DIM 128
#define E_DIM 300
#define LQ 128
#define LD 512
#define NB 128

// ---------------------------------------------------------------------------
// Kernel 1: repack conv weights (H, D, win) -> bf16 B-fragment tiles.
// B tile (K=32 x N=16): lane = (h&15) + 16*(kl>=16), slot = kl&15.
// Tiles ordered [nt][kt], kt-major contiguous per n-tile; K zero-padded.
// ---------------------------------------------------------------------------
__global__ void repack_wb_kernel(const float* __restrict__ w,
                                 unsigned short* __restrict__ wtb_u16,
                                 int win, int KT, int total) {
  int e = blockIdx.x * 256 + threadIdx.x;
  if (e >= total) return;
  int nt   = e / (KT * 512);
  int r    = e - nt * (KT * 512);
  int kt   = r >> 9;
  int lane = (r >> 4) & 31;
  int slot = r & 15;
  int h = nt * 16 + (lane & 15);
  int k = kt * 32 + slot + ((lane >= 16) ? 16 : 0);
  float v = 0.0f;
  int K = E_DIM * win;
  if (k < K) {
    int d  = k / win;
    int ww = k - d * win;
    v = w[(h * E_DIM + d) * win + ww];
  }
  ((__bf16*)wtb_u16)[e] = (__bf16)v;
}

// ---------------------------------------------------------------------------
// Kernel 2: WMMA conv: embedding gather + im2col + GEMM + tanh + L2-norm.
// Block = (b, 16-row l-tile). Build A pane in LDS (A-swizzle, bf16), then
// 8 waves each own h-tile nt and accumulate KT WMMAs (2 interleaved accums).
// Output swizzles (same as the match kernel expects):
// LAYOUT==0 (doc, A tiles): lane = l&15 + 16*((k>>3)&1), slot=(k&7)+8*(k>=16)
// LAYOUT==1 (query, B tiles): lane = l&15 + 16*((k>>3)&1), slot=(k&7)+8*(k>=16)
// ---------------------------------------------------------------------------
template <int WIN, int LAYOUT>
__global__ void __launch_bounds__(256)
conv_wmma_kernel(const int* __restrict__ tok, const float* __restrict__ emb,
                 int L, int nTiles, const unsigned short* __restrict__ wtb_u16,
                 const float* __restrict__ bias,
                 unsigned short* __restrict__ outp_u16) {
  constexpr int K  = E_DIM * WIN;
  constexpr int KT = (K + 31) / 32;   // k-tiles (zero-padded)
  constexpr int R  = 15 + WIN;        // embedding rows (right pad = WIN-1)
  __shared__ float sx[R * E_DIM];
  __shared__ __align__(1024) unsigned short sA_u16[KT * 512];
  __shared__ float snorm[16];

  const int b   = blockIdx.x / nTiles;
  const int lt  = blockIdx.x - b * nTiles;
  const int tid = threadIdx.x;
  const int l0  = lt * 16;

  // Stage f32 embedding rows (zero beyond L = right padding).
  for (int idx = tid; idx < R * E_DIM; idx += 256) {
    int r = idx / E_DIM, d = idx - r * E_DIM;
    int gl = l0 + r;
    float v = 0.0f;
    if (gl < L) v = emb[(size_t)tok[b * L + gl] * E_DIM + d];
    sx[idx] = v;
  }
  if (tid < 16) snorm[tid] = 0.0f;
  __syncthreads();

  // Build im2col A-fragment pane (bf16, A-swizzle). KT*512 is a multiple
  // of 256, so this loop is convergent.
  __bf16* sA = (__bf16*)sA_u16;
  for (int e = tid; e < KT * 512; e += 256) {
    int kt   = e >> 9;
    int lane = (e >> 4) & 31;
    int slot = e & 15;
    int lloc = lane & 15;
    int k = kt * 32 + slot + ((lane >= 16) ? 16 : 0);
    float v = 0.0f;
    if (k < K) {
      int d = k / WIN;
      int w = k - d * WIN;
      v = sx[(lloc + w) * E_DIM + d];
    }
    sA[e] = (__bf16)v;
  }
  __syncthreads();

  const int lane = tid & 31;
  const int nt   = tid >> 5;  // wave id = h-tile
  const __bf16* wtb = (const __bf16*)wtb_u16 + (size_t)nt * (KT * 512);

  v8f cc0 = {0.f, 0.f, 0.f, 0.f, 0.f, 0.f, 0.f, 0.f};
  v8f cc1 = {0.f, 0.f, 0.f, 0.f, 0.f, 0.f, 0.f, 0.f};
  int kt = 0;
  for (; kt + 1 < KT; kt += 2) {
    v16bf a0 = *(const v16bf*)(sA  + (kt << 9) + lane * 16);
    v16bf b0 = *(const v16bf*)(wtb + (kt << 9) + lane * 16);
    v16bf a1 = *(const v16bf*)(sA  + ((kt + 1) << 9) + lane * 16);
    v16bf b1 = *(const v16bf*)(wtb + ((kt + 1) << 9) + lane * 16);
    cc0 = __builtin_amdgcn_wmma_f32_16x16x32_bf16(false, a0, false, b0,
                                                  (short)0, cc0, false, false);
    cc1 = __builtin_amdgcn_wmma_f32_16x16x32_bf16(false, a1, false, b1,
                                                  (short)0, cc1, false, false);
  }
  if (KT & 1) {
    v16bf a0 = *(const v16bf*)(sA  + (kt << 9) + lane * 16);
    v16bf b0 = *(const v16bf*)(wtb + (kt << 9) + lane * 16);
    cc0 = __builtin_amdgcn_wmma_f32_16x16x32_bf16(false, a0, false, b0,
                                                  (short)0, cc0, false, false);
  }

  // Epilogue: C layout: VGPR v -> (M = v + 8*(lane>=16), N = lane&15).
  const int hbase = nt * 16 + (lane & 15);
  const float bv = bias[hbase];
  const int mofs = (lane >= 16) ? 8 : 0;
  float tv[8];
#pragma unroll
  for (int v = 0; v < 8; ++v) {
    tv[v] = tanhf(cc0[v] + cc1[v] + bv);
    atomicAdd(&snorm[v + mofs], tv[v] * tv[v]);
  }
  __syncthreads();

  __bf16* outp = (__bf16*)outp_u16 + (size_t)b * (size_t)L * H_DIM;
#pragma unroll
  for (int v = 0; v < 8; ++v) {
    int ml = v + mofs;
    float sc = 1.0f / fmaxf(sqrtf(snorm[ml]), 1e-12f);
    float ov = tv[v] * sc;
    int lglob = l0 + ml;
    int h = hbase;
    int k = h & 31;
    int idx;
    if (LAYOUT == 0) {
      int rt = lglob >> 4, ht = h >> 5;
      int olane = (lglob & 15) + (((k >> 3) & 1) << 4);
      int slot  = (k & 7) + ((k >> 4) << 3);
      idx = ((rt * 4 + ht) << 9) + olane * 16 + slot;
    } else {
      int ntq = lglob >> 4, ktq = h >> 5;
      int olane = (lglob & 15) + (((k >> 3) & 1) << 4);
      int slot  = (k & 7) + ((k >> 4) << 3);
      idx = ((ntq * 4 + ktq) << 9) + olane * 16 + slot;
    }
    outp[idx] = (__bf16)ov;
  }
}

// ---------------------------------------------------------------------------
// TDM helper: 1-D DMA of nElem8B * 8 bytes from global to LDS.
// D# per ISA ch.8: group0 = {count=1 | lds_addr | global_addr | type=2},
// group1 = {data_size=8B, tensor_dim0=tile_dim0=nElem8B, tensor_dim1=1,
//           tile_dim1=1, stride0=nElem8B}. Groups 2/3 zero (<=2D tensor).
// ---------------------------------------------------------------------------
__device__ __forceinline__ void tdm_copy_1d(unsigned lds_off,
                                            unsigned long long gaddr,
                                            int nElem8B) {
  v4u g0;
  g0.x = 1u;                                                   // count=1
  g0.y = (unsigned)__builtin_amdgcn_readfirstlane((int)lds_off);
  g0.z = (unsigned)__builtin_amdgcn_readfirstlane((int)(unsigned)(gaddr & 0xFFFFFFFFull));
  g0.w = ((unsigned)__builtin_amdgcn_readfirstlane((int)(unsigned)(gaddr >> 32)) & 0x01FFFFFFu)
         | 0x80000000u;                                        // type=2
  v8i g1;
  g1[0] = 3 << 16;                       // workgroup_mask=0, data_size=3 (8B)
  g1[1] = (nElem8B & 0xFFFF) << 16;      // tensor_dim0[15:0]  @ bits 63:48
  g1[2] = ((nElem8B >> 16) & 0xFFFF) | (1 << 16);  // dim0 hi | tensor_dim1=1
  g1[3] = (nElem8B & 0xFFFF) << 16;      // tile_dim0          @ bits 127:112
  g1[4] = 1;                             // tile_dim1=1, tile_dim2=0
  g1[5] = nElem8B;                       // tensor_dim0_stride lo
  g1[6] = 0;
  g1[7] = 0;
  v4i z4 = {0, 0, 0, 0};
#if __clang_major__ >= 23
  v8i z8 = {0, 0, 0, 0, 0, 0, 0, 0};
  __builtin_amdgcn_tensor_load_to_lds(g0, g1, z4, z4, z8, 0);
#else
  __builtin_amdgcn_tensor_load_to_lds(g0, g1, z4, z4, 0);
#endif
}

// ---------------------------------------------------------------------------
// Kernel 3: fused matching GEMM (bf16 WMMA) + Gaussian kernel pooling.
// One block per (b, qi, di). 8 waves; wave w owns query-column tile nt=w.
// Doc pane (512x128 bf16 = 128KB) DMA'd by wave 0 via TDM in 16KB chunks,
// double-buffered in LDS, shared by all 8 waves.
// ---------------------------------------------------------------------------
__global__ void __launch_bounds__(256)
match_pool_kernel(const unsigned short* __restrict__ qc_u16,
                  const unsigned short* __restrict__ dc_u16,
                  float* __restrict__ phi) {
  const int bid = blockIdx.x;  // b*9 + qi*3 + di
  const int b = bid / 9;
  const int rem = bid - b * 9;
  const int qi = rem / 3;
  const int di = rem - qi * 3;

  const __bf16* qb = (const __bf16*)qc_u16 + (size_t)(qi * NB + b) * (LQ * H_DIM);
  const __bf16* db = (const __bf16*)dc_u16 + (size_t)(di * NB + b) * (LD * H_DIM);

  const int lane = threadIdx.x & 31;
  const int wid  = threadIdx.x >> 5;  // wave id = query column tile nt

  __shared__ __align__(1024) unsigned char sAbuf[2][16384];
  __shared__ float sphi[11];

  // B fragments: reused across the whole doc loop.
  v16bf bfrag[4];
#pragma unroll
  for (int kt = 0; kt < 4; ++kt)
    bfrag[kt] = *(const v16bf*)(qb + ((wid * 4 + kt) << 9) + lane * 16);

  // mu_i = 0.1 + 0.2 i - 1; last bin clamps to (1.0, sigma=1e-3).
  const float MUS[11] = {-0.9f, -0.7f, -0.5f, -0.3f, -0.1f,
                         0.1f,  0.3f,  0.5f,  0.7f,  0.9f, 1.0f};
  // c2 = 1/(2 sigma^2) * log2(e) so exp(-t^2/2s^2) = exp2(-t^2*c2).
  const float C2[11] = {72.134752f, 72.134752f, 72.134752f, 72.134752f,
                        72.134752f, 72.134752f, 72.134752f, 72.134752f,
                        72.134752f, 72.134752f, 721347.52f};

  float acc[11];
#pragma unroll
  for (int i = 0; i < 11; ++i) acc[i] = 0.0f;

  const unsigned long long dga = (unsigned long long)(uintptr_t)db;
  const unsigned lds0 = (unsigned)(uintptr_t)(void*)&sAbuf[0][0];
  const unsigned lds1 = (unsigned)(uintptr_t)(void*)&sAbuf[1][0];

  if (wid == 0) tdm_copy_1d(lds0, dga, 2048);  // chunk 0 -> buf 0

  for (int c = 0; c < 8; ++c) {  // 8 chunks x 4 doc tiles = 32 doc tiles
    if (wid == 0) {
      if (c + 1 < 8) {
        tdm_copy_1d(((c + 1) & 1) ? lds1 : lds0,
                    dga + (unsigned long long)(c + 1) * 16384ull, 2048);
        __builtin_amdgcn_s_wait_tensorcnt(1);  // chunk c complete (in-order)
      } else {
        __builtin_amdgcn_s_wait_tensorcnt(0);
      }
    }
    __syncthreads();  // chunk c visible to all waves

    const __bf16* Abase = (const __bf16*)&sAbuf[c & 1][0];
    for (int ml = 0; ml < 4; ++ml) {
      v8f cc = {0.f, 0.f, 0.f, 0.f, 0.f, 0.f, 0.f, 0.f};
#pragma unroll
      for (int kt = 0; kt < 4; ++kt) {
        v16bf a = *(const v16bf*)(Abase + ((ml * 4 + kt) << 9) + lane * 16);
        cc = __builtin_amdgcn_wmma_f32_16x16x32_bf16(
            /*neg_a=*/false, a, /*neg_b=*/false, bfrag[kt],
            /*c_mod=*/(short)0, cc, /*reuse_a=*/false, /*reuse_b=*/false);
      }
      // Pool the 8 C elements of this lane into the 11 Gaussian bins.
#pragma unroll
      for (int v = 0; v < 8; ++v) {
        float x = cc[v];
#pragma unroll
        for (int i = 0; i < 11; ++i) {
          float t = x - MUS[i];
          acc[i] += __builtin_amdgcn_exp2f(-t * t * C2[i]);
        }
      }
    }
    __syncthreads();  // all waves done with buf[c&1] before it is re-filled
  }

  // Combine the two M-halves of each column, log1p, reduce over columns.
  float lp[11];
#pragma unroll
  for (int i = 0; i < 11; ++i) {
    float s = acc[i] + __shfl_xor(acc[i], 16, 32);  // full doc-axis sum
    float l2 = __builtin_amdgcn_logf(1.0f + s);     // log2(1+s)
    lp[i] = (lane < 16) ? l2 * 0.69314718055994531f : 0.0f;
    lp[i] += __shfl_xor(lp[i], 1, 32);
    lp[i] += __shfl_xor(lp[i], 2, 32);
    lp[i] += __shfl_xor(lp[i], 4, 32);
    lp[i] += __shfl_xor(lp[i], 8, 32);
  }

  if (threadIdx.x < 11) sphi[threadIdx.x] = 0.0f;
  __syncthreads();
  if (lane == 0) {
#pragma unroll
    for (int i = 0; i < 11; ++i) atomicAdd(&sphi[i], lp[i]);
  }
  __syncthreads();
  if (threadIdx.x < 11)
    phi[(size_t)b * 99 + (qi * 3 + di) * 11 + threadIdx.x] = sphi[threadIdx.x];
}

// ---------------------------------------------------------------------------
// Kernel 4: out[b] = phi[b,:] . out_w + out_b
// ---------------------------------------------------------------------------
__global__ void final_linear_kernel(const float* __restrict__ phi,
                                    const float* __restrict__ w,
                                    const float* __restrict__ b0,
                                    float* __restrict__ out) {
  int b = threadIdx.x;
  if (b < NB) {
    float s = b0[0];
#pragma unroll 3
    for (int j = 0; j < 99; ++j) s += phi[b * 99 + j] * w[j];
    out[b] = s;
  }
}

// ---------------------------------------------------------------------------
extern "C" void kernel_launch(void* const* d_in, const int* in_sizes, int n_in,
                              void* d_out, int out_size, void* d_ws,
                              size_t ws_size, hipStream_t stream) {
  const int*   query = (const int*)d_in[0];    // (128,128)
  const int*   doc   = (const int*)d_in[1];    // (128,512)
  const float* q_emb = (const float*)d_in[2];  // (30000,300)
  const float* d_emb = (const float*)d_in[3];  // (30000,300)
  const float* out_w = (const float*)d_in[4];  // (1,99)
  const float* out_b = (const float*)d_in[5];  // (1,)
  const float* conv_w[3] = {(const float*)d_in[6], (const float*)d_in[8],
                            (const float*)d_in[10]};
  const float* conv_b[3] = {(const float*)d_in[7], (const float*)d_in[9],
                            (const float*)d_in[11]};

  // Workspace layout (bytes):
  char* ws = (char*)d_ws;
  const int KTs[3] = {19, 29, 38};  // ceil(300*win/32), win = 2,3,4
  unsigned short* wtb[3] = {(unsigned short*)(ws + 0),
                            (unsigned short*)(ws + 155648),
                            (unsigned short*)(ws + 393216)};
  unsigned short* qc = (unsigned short*)(ws + 1048576);   // 3*128*128*128 bf16
  unsigned short* dc = (unsigned short*)(ws + 13631488);  // 3*128*512*128 bf16
  float* phi = (float*)(ws + 63963136);                   // 128*99 f32
  (void)ws_size;  // requires ~64.1 MB of workspace

  // 1) Repack weights into bf16 B-fragment tiles (K zero-padded).
  for (int ci = 0; ci < 3; ++ci) {
    int win = ci + 2;
    int total = 8 * KTs[ci] * 512;
    repack_wb_kernel<<<(total + 255) / 256, 256, 0, stream>>>(
        conv_w[ci], wtb[ci], win, KTs[ci], total);
  }

  // 2) WMMA conv + tanh + l2norm, fragment-swizzled bf16 outputs.
  conv_wmma_kernel<2, 1><<<NB * (LQ / 16), 256, 0, stream>>>(
      query, q_emb, LQ, LQ / 16, wtb[0], conv_b[0], qc + (size_t)0 * NB * LQ * H_DIM);
  conv_wmma_kernel<3, 1><<<NB * (LQ / 16), 256, 0, stream>>>(
      query, q_emb, LQ, LQ / 16, wtb[1], conv_b[1], qc + (size_t)1 * NB * LQ * H_DIM);
  conv_wmma_kernel<4, 1><<<NB * (LQ / 16), 256, 0, stream>>>(
      query, q_emb, LQ, LQ / 16, wtb[2], conv_b[2], qc + (size_t)2 * NB * LQ * H_DIM);
  conv_wmma_kernel<2, 0><<<NB * (LD / 16), 256, 0, stream>>>(
      doc, d_emb, LD, LD / 16, wtb[0], conv_b[0], dc + (size_t)0 * NB * LD * H_DIM);
  conv_wmma_kernel<3, 0><<<NB * (LD / 16), 256, 0, stream>>>(
      doc, d_emb, LD, LD / 16, wtb[1], conv_b[1], dc + (size_t)1 * NB * LD * H_DIM);
  conv_wmma_kernel<4, 0><<<NB * (LD / 16), 256, 0, stream>>>(
      doc, d_emb, LD, LD / 16, wtb[2], conv_b[2], dc + (size_t)2 * NB * LD * H_DIM);

  // 3) Fused WMMA matching + Gaussian pooling: one block per (b, qi, di).
  match_pool_kernel<<<NB * 9, 256, 0, stream>>>(qc, dc, phi);

  // 4) Final linear.
  final_linear_kernel<<<1, 128, 0, stream>>>(phi, out_w, out_b, (float*)d_out);
}
